// M13_5514738008552
// MI455X (gfx1250) — matrix operand order repaired and checked by reading, not compiled, not run on hardware
//
#include <hip/hip_runtime.h>
#include <math.h>

typedef __attribute__((ext_vector_type(16))) _Float16 v16h;
typedef __attribute__((ext_vector_type(8)))  _Float16 v8h;
typedef __attribute__((ext_vector_type(8)))  float    v8f;

#define NEG_SLOPE 0.01f
#define BN_EPSF   1e-5f

// A-loader modes for the fused GEMM
#define GEMM_PLAIN 0   // A = A0[M,K]
#define GEMM_GINE  1   // A = A0 + (1+eps)*A1   (agg + (1+eps)*h), both [M,K]
#define GEMM_CAT   2   // A = [A0 (M,256) | A1 (M,64)], K = 320

// ---------------------------------------------------------------------------
// GINEConv edge stage: one wave = 16-edge x 16-col tile of
//   e = edge_attr(16x16) @ eW(16 x in_c)   (one v_wmma_f32_16x16x32_f16,
// K=16 zero-padded to 32), then per element:
//   m = relu(e + eb[n] + h[src[e]][n]);  atomicAdd(agg[dst[e]][n], m)
// ---------------------------------------------------------------------------
__global__ void edge_gine_kernel(const float* __restrict__ edge_attr, // [E,16]
                                 const int*   __restrict__ src,       // [E]
                                 const int*   __restrict__ dst,       // [E]
                                 const float* __restrict__ eW,        // [16,in_c]
                                 const float* __restrict__ eb,        // [in_c]
                                 const float* __restrict__ h,         // [N,in_c]
                                 float*       __restrict__ agg,       // [N,in_c]
                                 int E, int in_c)
{
    const int ctiles      = in_c >> 4;
    const int tiles_total = (E >> 4) * ctiles;
    const int wave = threadIdx.x >> 5;
    const int tile = blockIdx.x * (blockDim.x >> 5) + wave;
    if (tile >= tiles_total) return;           // wave-uniform: EXEC all-1 at WMMA
    const int etile = tile / ctiles;
    const int ctile = tile - etile * ctiles;

    const int lane = threadIdx.x & 31;
    const int half = lane >> 4;
    const int l16  = lane & 15;

    // A fragment (16x32 f16 layout): lane row M=l16; i<8 -> K=8*half+i (valid),
    // i>=8 -> K>=16 -> zero pad. Contiguous K runs => two float4 loads.
    const float* arow = edge_attr + (unsigned)(etile * 16 + l16) * 16u;
    float4 alo = ((const float4*)arow)[2 * half];
    float4 ahi = ((const float4*)arow)[2 * half + 1];
    v16h a;
    a[0] = (_Float16)alo.x; a[1] = (_Float16)alo.y;
    a[2] = (_Float16)alo.z; a[3] = (_Float16)alo.w;
    a[4] = (_Float16)ahi.x; a[5] = (_Float16)ahi.y;
    a[6] = (_Float16)ahi.z; a[7] = (_Float16)ahi.w;
    #pragma unroll
    for (int i = 8; i < 16; ++i) a[i] = (_Float16)0.0f;

    // B fragment (32x16 f16): lane column N; K = 8*half + i (i<8), 16 + 8*half + (i-8) (i>=8), pad K>=16 -> 0
    v16h b;
    const int ncol = ctile * 16 + l16;
    #pragma unroll
    for (int i = 0; i < 16; ++i) {
        int k = (i < 8) ? (8 * half + i) : (16 + 8 * half + (i - 8));
        b[i] = (k < 16) ? (_Float16)eW[k * in_c + ncol] : (_Float16)0.0f;
    }
    v8f acc = {};
    acc = __builtin_amdgcn_wmma_f32_16x16x32_f16(false, a, false, b,
                                                 (short)0, acc, false, false);

    const float bias = eb[ncol];
    #pragma unroll
    for (int r = 0; r < 8; ++r) {
        int m = r + 8 * half;                  // C/D: VGPR r -> M = r + 8*half
        int e = etile * 16 + m;
        unsigned s = (unsigned)src[e];
        unsigned d = (unsigned)dst[e];
        float v = acc[r] + bias + h[s * (unsigned)in_c + (unsigned)ncol];
        v = v > 0.0f ? v : 0.0f;
        atomicAdd(&agg[d * (unsigned)in_c + (unsigned)ncol], v);
    }
}

// Load 8 consecutive A elements starting at column kg (kg multiple of 8)
__device__ __forceinline__ void grp8(float4& lo, float4& hi,
                                     const float* r0, const float* r1,
                                     int mode, float eps1, int kg)
{
    const float* p = r0 + kg;
    if (mode == GEMM_CAT && kg >= 256) p = r1 + (kg - 256);
    lo = ((const float4*)p)[0];
    hi = ((const float4*)p)[1];
    if (mode == GEMM_GINE) {
        float4 hlo = ((const float4*)(r1 + kg))[0];
        float4 hhi = ((const float4*)(r1 + kg))[1];
        lo.x += eps1 * hlo.x; lo.y += eps1 * hlo.y;
        lo.z += eps1 * hlo.z; lo.w += eps1 * hlo.w;
        hi.x += eps1 * hhi.x; hi.y += eps1 * hhi.y;
        hi.z += eps1 * hhi.z; hi.w += eps1 * hhi.w;
    }
}

__device__ __forceinline__ v16h ldsB(const _Float16 (*Wt)[40], int n, int half)
{
    v8h lo = *(const v8h*)&Wt[n][8 * half];       // 16B-aligned ds_load_b128
    v8h hi = *(const v8h*)&Wt[n][16 + 8 * half];
    v16h b;
    #pragma unroll
    for (int i = 0; i < 8; ++i) { b[i] = lo[i]; b[i + 8] = hi[i]; }
    return b;
}

// ---------------------------------------------------------------------------
// Fused WMMA GEMM: out[M,Nc] = Aload(mode)[M,K] @ W[K,Nc] + bias
// Block = 128 threads = 4 waves => 64x64 output tile.
// Per K-step (32): stage W panel (32x64) into LDS as transposed f16 (row pitch
// 40 halfs => conflict-free), each wave reuses one A fragment for 4 WMMAs.
// M multiple of 16, K multiple of 32, Nc multiple of 64 at every call site.
// ---------------------------------------------------------------------------
__global__ void gemm_wmma_kernel(const float* __restrict__ A0,
                                 const float* __restrict__ A1,
                                 const float* __restrict__ epsp,
                                 const float* __restrict__ W,   // [K,Nc]
                                 const float* __restrict__ bias,
                                 float* __restrict__ out,
                                 int M, int K, int Nc, int mode)
{
    __shared__ _Float16 Wt[64][40];   // [n_local][k_local], padded pitch

    const int wave = threadIdx.x >> 5;
    const int lane = threadIdx.x & 31;
    const int half = lane >> 4;
    const int l16  = lane & 15;

    const int mtile   = blockIdx.y * 4 + wave;
    const int colBase = blockIdx.x * 64;
    const bool active = (mtile * 16) < M;

    const float eps1 = (mode == GEMM_GINE) ? (1.0f + epsp[0]) : 0.0f;

    const float* r0 = A0;
    const float* r1 = A1;
    if (active) {
        const unsigned row = (unsigned)(mtile * 16 + l16);
        if (mode == GEMM_CAT) { r0 = A0 + (size_t)row * 256; r1 = A1 + (size_t)row * 64; }
        else {
            r0 = A0 + (size_t)row * (unsigned)K;
            if (mode == GEMM_GINE) r1 = A1 + (size_t)row * (unsigned)K;
        }
    }

    v8f acc0 = {}, acc1 = {}, acc2 = {}, acc3 = {};

    for (int kb = 0; kb < K; kb += 32) {
        __syncthreads();   // previous readers done before restaging
        // ---- cooperative stage: W[kb..kb+32) x [colBase..+64) -> Wt^T (f16) ----
        #pragma unroll
        for (int it = 0; it < 2; ++it) {
            int g   = threadIdx.x + it * 128;      // 256 groups of 8
            int kk  = g >> 3;
            int nn0 = (g & 7) << 3;
            const float* wp = W + (size_t)(unsigned)(kb + kk) * (unsigned)Nc
                                + (unsigned)(colBase + nn0);
            float4 w0 = ((const float4*)wp)[0];
            float4 w1 = ((const float4*)wp)[1];
            Wt[nn0 + 0][kk] = (_Float16)w0.x;
            Wt[nn0 + 1][kk] = (_Float16)w0.y;
            Wt[nn0 + 2][kk] = (_Float16)w0.z;
            Wt[nn0 + 3][kk] = (_Float16)w0.w;
            Wt[nn0 + 4][kk] = (_Float16)w1.x;
            Wt[nn0 + 5][kk] = (_Float16)w1.y;
            Wt[nn0 + 6][kk] = (_Float16)w1.z;
            Wt[nn0 + 7][kk] = (_Float16)w1.w;
        }
        __syncthreads();

        if (active) {
            // A fragment: K groups [kb+8*half, +8) and [kb+16+8*half, +8)
            float4 l0, h0, l1, h1;
            grp8(l0, h0, r0, r1, mode, eps1, kb + 8 * half);
            grp8(l1, h1, r0, r1, mode, eps1, kb + 16 + 8 * half);
            v16h a;
            a[0]  = (_Float16)l0.x; a[1]  = (_Float16)l0.y;
            a[2]  = (_Float16)l0.z; a[3]  = (_Float16)l0.w;
            a[4]  = (_Float16)h0.x; a[5]  = (_Float16)h0.y;
            a[6]  = (_Float16)h0.z; a[7]  = (_Float16)h0.w;
            a[8]  = (_Float16)l1.x; a[9]  = (_Float16)l1.y;
            a[10] = (_Float16)l1.z; a[11] = (_Float16)l1.w;
            a[12] = (_Float16)h1.x; a[13] = (_Float16)h1.y;
            a[14] = (_Float16)h1.z; a[15] = (_Float16)h1.w;

            v16h b0 = ldsB(Wt,  0 + l16, half);
            v16h b1 = ldsB(Wt, 16 + l16, half);
            v16h b2 = ldsB(Wt, 32 + l16, half);
            v16h b3 = ldsB(Wt, 48 + l16, half);
            acc0 = __builtin_amdgcn_wmma_f32_16x16x32_f16(false, a, false, b0,
                                                          (short)0, acc0, false, false);
            acc1 = __builtin_amdgcn_wmma_f32_16x16x32_f16(false, a, false, b1,
                                                          (short)0, acc1, false, false);
            acc2 = __builtin_amdgcn_wmma_f32_16x16x32_f16(false, a, false, b2,
                                                          (short)0, acc2, false, false);
            acc3 = __builtin_amdgcn_wmma_f32_16x16x32_f16(false, a, false, b3,
                                                          (short)0, acc3, false, false);
        }
    }

    if (active) {
        const float bv0 = bias[colBase + l16];
        const float bv1 = bias[colBase + 16 + l16];
        const float bv2 = bias[colBase + 32 + l16];
        const float bv3 = bias[colBase + 48 + l16];
        #pragma unroll
        for (int r = 0; r < 8; ++r) {
            int m = mtile * 16 + r + 8 * half;
            float* orow = out + (size_t)(unsigned)m * (unsigned)Nc
                              + (unsigned)(colBase + l16);
            orow[0]  = acc0[r] + bv0;
            orow[16] = acc1[r] + bv1;
            orow[32] = acc2[r] + bv2;
            orow[48] = acc3[r] + bv3;
        }
    }
}

// Column-wise BatchNorm stats (biased var): one block per column, LDS tree.
__global__ void bn_stats_kernel(const float* __restrict__ X, int M, int Nc,
                                float* __restrict__ mu, float* __restrict__ istd)
{
    __shared__ float ssum[256];
    __shared__ float ssq[256];
    const int c = blockIdx.x;
    float s = 0.0f, q = 0.0f;
    for (int r = threadIdx.x; r < M; r += blockDim.x) {
        float v = X[(size_t)(unsigned)r * (unsigned)Nc + (unsigned)c];
        s += v; q += v * v;
    }
    ssum[threadIdx.x] = s; ssq[threadIdx.x] = q;
    __syncthreads();
    for (int off = blockDim.x >> 1; off > 0; off >>= 1) {
        if (threadIdx.x < off) {
            ssum[threadIdx.x] += ssum[threadIdx.x + off];
            ssq[threadIdx.x]  += ssq[threadIdx.x + off];
        }
        __syncthreads();
    }
    if (threadIdx.x == 0) {
        float m   = ssum[0] / (float)M;
        float var = ssq[0] / (float)M - m * m;
        mu[c]   = m;
        istd[c] = rsqrtf(var + BN_EPSF);
    }
}

// y = leaky_relu((x - mu) * istd * g + b), in place
__global__ void bn_apply_leaky_kernel(float* __restrict__ X,
                                      const float* __restrict__ mu,
                                      const float* __restrict__ istd,
                                      const float* __restrict__ g,
                                      const float* __restrict__ be,
                                      int total, int Nc)
{
    int i = blockIdx.x * blockDim.x + threadIdx.x;
    if (i >= total) return;
    int c = i & (Nc - 1);                 // Nc is 256 or 128 (power of two)
    float v = (X[i] - mu[c]) * istd[c] * g[c] + be[c];
    X[i] = v > 0.0f ? v : NEG_SLOPE * v;
}

// out[n] = dot(Z[n, :128], w) + b
__global__ void final_dot_kernel(const float* __restrict__ Z,
                                 const float* __restrict__ w,
                                 const float* __restrict__ b,
                                 float* __restrict__ out, int N)
{
    int n = blockIdx.x * blockDim.x + threadIdx.x;
    if (n >= N) return;
    const float* row = Z + (size_t)(unsigned)n * 128u;
    float s = 0.0f;
    #pragma unroll 8
    for (int k = 0; k < 128; ++k) s += row[k] * w[k];
    out[n] = s + b[0];
}

// ---------------------------------------------------------------------------
// d_in layout (setup_inputs order, flattened):
//  0:x  1:edge_index  2:edge_attr  3:mol_x
//  4+9l : conv layer l {eps,eW,eb,W1,b1,g1,be1,W2,b2}  (l=0..2)
//  31..34 : bn_params {g0,b0,g1,b1}
//  35..40 : final {W1,b1,g,be,W2,b2}
// ---------------------------------------------------------------------------
extern "C" void kernel_launch(void* const* d_in, const int* in_sizes, int n_in,
                              void* d_out, int out_size, void* d_ws, size_t ws_size,
                              hipStream_t stream)
{
    (void)in_sizes; (void)n_in; (void)out_size; (void)ws_size;
    const int N = 50000, E = 800000, Cc = 256, FNODE = 128, FMOL = 64, FFIN = 128;

    const float* x         = (const float*)d_in[0];
    const int*   ei        = (const int*)d_in[1];   // [2,E]: row0=src, row1=dst
    const float* edge_attr = (const float*)d_in[2];
    const float* mol_x     = (const float*)d_in[3];
    const int* src = ei;
    const int* dst = ei + E;

    // workspace carve (floats): 4 x [N,256] + stats
    float* bufA = (float*)d_ws;
    float* bufB = bufA + (size_t)N * Cc;
    float* bufC = bufB + (size_t)N * Cc;
    float* bufD = bufC + (size_t)N * Cc;
    float* mu   = bufD + (size_t)N * Cc;
    float* istd = mu + Cc;

    const int Mtiles  = N / 16;                    // 3125
    const int gemmGy  = (Mtiles + 3) / 4;          // 782 (4 waves per block)

    const float* h = x;
    float* houts[3] = { bufA, bufD, bufA };        // ping-pong; A free again by l=2

    for (int l = 0; l < 3; ++l) {
        const int base = 4 + 9 * l;
        const float* epsp = (const float*)d_in[base + 0];
        const float* eW   = (const float*)d_in[base + 1];
        const float* eb   = (const float*)d_in[base + 2];
        const float* W1   = (const float*)d_in[base + 3];
        const float* b1   = (const float*)d_in[base + 4];
        const float* g1   = (const float*)d_in[base + 5];
        const float* be1  = (const float*)d_in[base + 6];
        const float* W2   = (const float*)d_in[base + 7];
        const float* b2   = (const float*)d_in[base + 8];
        const int in_c = (l == 0) ? FNODE : Cc;

        hipMemsetAsync(bufB, 0, (size_t)N * in_c * sizeof(float), stream);

        {   // edge messages + aggregation into bufB
            int tiles  = (E / 16) * (in_c / 16);
            int blocks = (tiles + 3) / 4;
            edge_gine_kernel<<<blocks, 128, 0, stream>>>(edge_attr, src, dst, eW, eb,
                                                         h, bufB, E, in_c);
        }
        {   // GEMM1 (fused z = agg + (1+eps)*h): -> bufC [N,256]
            dim3 grid(Cc / 64, gemmGy);
            gemm_wmma_kernel<<<grid, 128, 0, stream>>>(bufB, h, epsp, W1, b1, bufC,
                                                       N, in_c, Cc, GEMM_GINE);
        }
        bn_stats_kernel<<<Cc, 256, 0, stream>>>(bufC, N, Cc, mu, istd);
        bn_apply_leaky_kernel<<<(N * Cc + 255) / 256, 256, 0, stream>>>(
            bufC, mu, istd, g1, be1, N * Cc, Cc);
        {   // GEMM2: bufC @ W2 -> h_out [N,256]
            dim3 grid(Cc / 64, gemmGy);
            gemm_wmma_kernel<<<grid, 128, 0, stream>>>(bufC, nullptr, nullptr, W2, b2,
                                                       houts[l], N, Cc, Cc, GEMM_PLAIN);
        }
        if (l < 2) {    // outer BN + leaky for all but last conv layer
            const float* bg = (const float*)d_in[31 + 2 * l];
            const float* bb = (const float*)d_in[31 + 2 * l + 1];
            bn_stats_kernel<<<Cc, 256, 0, stream>>>(houts[l], N, Cc, mu, istd);
            bn_apply_leaky_kernel<<<(N * Cc + 255) / 256, 256, 0, stream>>>(
                houts[l], mu, istd, bg, bb, N * Cc, Cc);
        }
        h = houts[l];
    }

    // ----- final MLP (concat fused into GEMM A-loader) -----
    const float* fW1 = (const float*)d_in[35];
    const float* fb1 = (const float*)d_in[36];
    const float* fg  = (const float*)d_in[37];
    const float* fbe = (const float*)d_in[38];
    const float* fW2 = (const float*)d_in[39];
    const float* fb2 = (const float*)d_in[40];

    {   // [h | mol_x] (K=320) @ fW1 -> bufC [N,128]
        dim3 grid(FFIN / 64, gemmGy);
        gemm_wmma_kernel<<<grid, 128, 0, stream>>>(h, mol_x, nullptr, fW1, fb1, bufC,
                                                   N, Cc + FMOL, FFIN, GEMM_CAT);
    }
    bn_stats_kernel<<<FFIN, 256, 0, stream>>>(bufC, N, FFIN, mu, istd);
    bn_apply_leaky_kernel<<<(N * FFIN + 255) / 256, 256, 0, stream>>>(
        bufC, mu, istd, fg, fbe, N * FFIN, FFIN);
    final_dot_kernel<<<(N + 255) / 256, 256, 0, stream>>>(bufC, fW2, fb2,
                                                          (float*)d_out, N);
}
